// GraphAttentionConv_27831388078879
// MI455X (gfx1250) — matrix operand order repaired and checked by reading, not compiled, not run on hardware
//
#include <hip/hip_runtime.h>

#define N_NODES 4096
#define UNITS 64
#define FEAT 64
#define BATCH 4
#define LRELU_ALPHA 0.2f

typedef __attribute__((ext_vector_type(16))) _Float16 v16h;
typedef __attribute__((ext_vector_type(8)))  float    v8f;

// ---------------------------------------------------------------------------
// Pass 1: Wh = x @ W (f32 in LDS), emit WhT (f16, [b][u][n]) + e_src/e_dst.
// One block = 64 nodes of one batch. 256 threads.
// ---------------------------------------------------------------------------
__global__ __launch_bounds__(256) void gat_wh_kernel(
    const float* __restrict__ x, const float* __restrict__ W,
    const float* __restrict__ asrc, const float* __restrict__ adst,
    _Float16* __restrict__ WhT, float* __restrict__ es, float* __restrict__ ed)
{
  __shared__ float xs[64][65];   // padded: kill bank conflicts
  __shared__ float ks[64][65];
  __shared__ float whs[64][65];
  const int tid = threadIdx.x;
  const int b  = blockIdx.x / (N_NODES / 64);
  const int n0 = (blockIdx.x % (N_NODES / 64)) * 64;

  for (int i = tid; i < 64 * 64; i += 256) {
    int r = i >> 6, c = i & 63;
    xs[r][c] = x[((size_t)b * N_NODES + n0 + r) * FEAT + c];
    ks[r][c] = W[i];                 // W is [FEAT][UNITS] row-major
  }
  __syncthreads();

  const int u  = tid & 63;
  const int nb = tid >> 6;
  #pragma unroll
  for (int i = 0; i < 16; ++i) {
    int node = nb * 16 + i;
    float acc = 0.f;
    #pragma unroll
    for (int f = 0; f < FEAT; ++f) acc += xs[node][f] * ks[f][u];
    whs[node][u] = acc;
  }
  __syncthreads();

  if (tid < 128) {                   // e_src (threads 0-63), e_dst (64-127)
    int node = tid & 63;
    const float* av = (tid < 64) ? asrc : adst;
    float s = 0.f;
    #pragma unroll
    for (int uu = 0; uu < UNITS; ++uu) s += whs[node][uu] * av[uu];
    float* dst = (tid < 64) ? es : ed;
    dst[(size_t)b * N_NODES + n0 + node] = s;
  }

  for (int i = tid; i < 64 * 64; i += 256) {       // coalesced along node dim
    int uu = i >> 6, node = i & 63;
    WhT[((size_t)b * UNITS + uu) * N_NODES + n0 + node] = (_Float16)whs[node][uu];
  }
}

// ---------------------------------------------------------------------------
// Pass 2: per-row softmax stats m[b,i] and 1/Z[b,i]. One block per (b,i).
// Branch-free: masked logits become -1e30 and their exp flushes to 0.
// ---------------------------------------------------------------------------
__global__ __launch_bounds__(256) void gat_stats_kernel(
    const float* __restrict__ adj, const float* __restrict__ es,
    const float* __restrict__ ed, float* __restrict__ mrow,
    float* __restrict__ rz)
{
  __shared__ float red[256];
  const int tid = threadIdx.x;
  const int b = blockIdx.x & (BATCH - 1);
  const int i = blockIdx.x >> 2;
  const float esi = es[(size_t)b * N_NODES + i];
  const float* adjr = adj + (size_t)i * N_NODES;
  const float* edb  = ed  + (size_t)b * N_NODES;

  float ev[16];
  float lmax = -1e30f;
  #pragma unroll
  for (int k = 0; k < 16; ++k) {
    int j = tid + 256 * k;
    float a = adjr[j];
    float e = esi + edb[j];
    e = fmaxf(e, LRELU_ALPHA * e);           // leaky relu (slope < 1)
    e = (a > 0.f) ? e : -1e30f;              // adjacency mask, one cndmask
    ev[k] = e;
    lmax = fmaxf(lmax, e);
  }
  red[tid] = lmax; __syncthreads();
  for (int s = 128; s > 0; s >>= 1) {
    if (tid < s) red[tid] = fmaxf(red[tid], red[tid + s]);
    __syncthreads();
  }
  const float m = red[0];
  __syncthreads();

  float lsum = 0.f;
  #pragma unroll
  for (int k = 0; k < 16; ++k)
    lsum += __expf(ev[k] - m);               // masked -> exp(-huge) == 0
  red[tid] = lsum; __syncthreads();
  for (int s = 128; s > 0; s >>= 1) {
    if (tid < s) red[tid] += red[tid + s];
    __syncthreads();
  }
  if (tid == 0) {
    mrow[(size_t)b * N_NODES + i] = m;
    rz[(size_t)b * N_NODES + i]  = 1.f / red[0];
  }
}

// ---------------------------------------------------------------------------
// Pass 3: h' = softmax(e) @ Wh via v_wmma_f32_16x16x32_f16.
// One block = one (b, 16-row tile). 4 waves split the j (K) range with a
// compile-time trip count (no divergent loop guard -> plain scalar loop).
// A holds p_ij = exp(e_ij - m_i) (mask folded into the logit, branch-free);
// the 1/Z_i row scale is applied once in the epilogue.
// ---------------------------------------------------------------------------
__global__ __launch_bounds__(128) void gat_attn_kernel(
    const float* __restrict__ adj, const float* __restrict__ es,
    const float* __restrict__ ed, const float* __restrict__ mrow,
    const float* __restrict__ rz, const _Float16* __restrict__ WhT,
    const float* __restrict__ bias, float* __restrict__ out)
{
  __shared__ float accs[4][16 * UNITS];
  const int tid  = threadIdx.x;
  const int wave = tid >> 5;
  const int lane = tid & 31;
  const int b  = blockIdx.x & (BATCH - 1);
  const int i0 = (blockIdx.x >> 2) << 4;

  const int M   = lane & 15;
  const int hi  = lane >> 4;           // 0: lanes 0-15, 1: lanes 16-31
  const int row = i0 + M;
  const float es_r = es[(size_t)b * N_NODES + row];
  const float m_r  = mrow[(size_t)b * N_NODES + row];

  const int kbase = hi * 16;           // A-matrix K base for this lane half
  const int koffB = hi * 16;           // B-matrix K base for this lane half
  const float* adjr = adj + (size_t)row * N_NODES;
  const float* edb  = ed  + (size_t)b * N_NODES;
  const _Float16* whb = WhT + (size_t)b * UNITS * N_NODES;
  const int n = lane & 15;             // B/C column index

  v8f c0 = {}, c1 = {}, c2 = {}, c3 = {};

  const int jb = wave * 32;            // this wave's base j offset
  #define J_ITERS (N_NODES / 128)      // 32 iterations, compile-time constant

  for (int it = 0; it < J_ITERS; ++it) {
    const int j0 = jb + it * 128;
    // lane<16 needs K={0..15}, lane>=16 needs K={16..31}
    const float4 e0 = *(const float4*)(edb + j0 + kbase);
    const float4 e1 = *(const float4*)(edb + j0 + kbase + 4);
    const float4 e2 = *(const float4*)(edb + j0 + kbase + 8);
    const float4 e3 = *(const float4*)(edb + j0 + kbase + 12);
    const float4 a0 = *(const float4*)(adjr + j0 + kbase);
    const float4 a1 = *(const float4*)(adjr + j0 + kbase + 4);
    const float4 a2 = *(const float4*)(adjr + j0 + kbase + 8);
    const float4 a3 = *(const float4*)(adjr + j0 + kbase + 12);
    float ee[16] = {e0.x, e0.y, e0.z, e0.w, e1.x, e1.y, e1.z, e1.w,
                    e2.x, e2.y, e2.z, e2.w, e3.x, e3.y, e3.z, e3.w};
    float aa[16] = {a0.x, a0.y, a0.z, a0.w, a1.x, a1.y, a1.z, a1.w,
                    a2.x, a2.y, a2.z, a2.w, a3.x, a3.y, a3.z, a3.w};

    v16h A;
    #pragma unroll
    for (int q = 0; q < 16; ++q) {
      float e = es_r + ee[q];
      e = fmaxf(e, LRELU_ALPHA * e);           // leaky relu
      e = (aa[q] > 0.f) ? e : -1e30f;          // mask -> exp gives exact 0
      A[q] = (_Float16)__expf(e - m_r);        // unconditional, no branch
    }

    // B tiles: lane holds column u=16t+n, 16 consecutive K values (32B load)
    const v16h B0 = *(const v16h*)(whb + ((size_t)(0 * 16 + n)) * N_NODES + j0 + koffB);
    const v16h B1 = *(const v16h*)(whb + ((size_t)(1 * 16 + n)) * N_NODES + j0 + koffB);
    const v16h B2 = *(const v16h*)(whb + ((size_t)(2 * 16 + n)) * N_NODES + j0 + koffB);
    const v16h B3 = *(const v16h*)(whb + ((size_t)(3 * 16 + n)) * N_NODES + j0 + koffB);

    c0 = __builtin_amdgcn_wmma_f32_16x16x32_f16(false, A, false, B0, (short)0, c0, false, false);
    c1 = __builtin_amdgcn_wmma_f32_16x16x32_f16(false, A, false, B1, (short)0, c1, false, false);
    c2 = __builtin_amdgcn_wmma_f32_16x16x32_f16(false, A, false, B2, (short)0, c2, false, false);
    c3 = __builtin_amdgcn_wmma_f32_16x16x32_f16(false, A, false, B3, (short)0, c3, false, false);
  }

  // C layout: VGPR r -> M = r + 8*hi, column = lane&15
  const int mbase = hi * 8;
  #pragma unroll
  for (int r = 0; r < 8; ++r) {
    accs[wave][(mbase + r) * UNITS + 0 * 16 + n] = c0[r];
    accs[wave][(mbase + r) * UNITS + 1 * 16 + n] = c1[r];
    accs[wave][(mbase + r) * UNITS + 2 * 16 + n] = c2[r];
    accs[wave][(mbase + r) * UNITS + 3 * 16 + n] = c3[r];
  }
  __syncthreads();

  for (int idx = tid; idx < 16 * UNITS; idx += 128) {
    float s = accs[0][idx] + accs[1][idx] + accs[2][idx] + accs[3][idx];
    int mm = idx >> 6;
    int uu = idx & 63;
    // apply the per-row 1/Z here instead of per element in the hot loop
    float scale = rz[(size_t)b * N_NODES + i0 + mm];
    out[((size_t)b * N_NODES + i0 + mm) * UNITS + uu] = s * scale + bias[uu];
  }
}

// ---------------------------------------------------------------------------
extern "C" void kernel_launch(void* const* d_in, const int* in_sizes, int n_in,
                              void* d_out, int out_size, void* d_ws, size_t ws_size,
                              hipStream_t stream)
{
  const float* x    = (const float*)d_in[0];
  const float* W    = (const float*)d_in[1];
  const float* asrc = (const float*)d_in[2];
  const float* adst = (const float*)d_in[3];
  const float* bias = (const float*)d_in[4];
  const float* adj  = (const float*)d_in[5];
  float* out = (float*)d_out;

  char* ws = (char*)d_ws;
  _Float16* WhT = (_Float16*)ws;                               // 2 MB, 256B aligned
  size_t off = (size_t)BATCH * UNITS * N_NODES * sizeof(_Float16);
  float* es   = (float*)(ws + off); off += (size_t)BATCH * N_NODES * sizeof(float);
  float* ed   = (float*)(ws + off); off += (size_t)BATCH * N_NODES * sizeof(float);
  float* mrow = (float*)(ws + off); off += (size_t)BATCH * N_NODES * sizeof(float);
  float* rz   = (float*)(ws + off);

  gat_wh_kernel<<<BATCH * (N_NODES / 64), 256, 0, stream>>>(x, W, asrc, adst, WhT, es, ed);
  gat_stats_kernel<<<BATCH * N_NODES, 256, 0, stream>>>(adj, es, ed, mrow, rz);
  gat_attn_kernel<<<BATCH * (N_NODES / 16), 128, 0, stream>>>(adj, es, ed, mrow, rz, WhT, bias, out);
}
